// ViTMoE_78529182040633
// MI455X (gfx1250) — matrix operand order repaired and checked by reading, not compiled, not run on hardware
//
#include <hip/hip_runtime.h>
#include <hip/hip_bf16.h>
#include <math.h>

// ---------------------------------------------------------------------------
// ViT-MoE forward for MI455X (gfx1250, wave32, WMMA).
// All dense GEMMs run on v_wmma_f32_16x16x32_f16 (f16 A/B, f32 acc) with
// pre-transposed f16 weights, double-buffered LDS, 1 barrier per K-step.
// ---------------------------------------------------------------------------

#define LL 4
#define EE 12
#define CC 256
#define MM 128
#define BB 128
#define PP 196        // 14*14 tokens
#define LDSK 48       // LDS K stride in halves (96B, keeps 16B alignment)

typedef __attribute__((ext_vector_type(16))) _Float16 v16h;
typedef __attribute__((ext_vector_type(8)))  _Float16 v8h;
typedef __attribute__((ext_vector_type(8)))  float    v8f;

// Build a 16-half fragment from two 16B LDS chunks.
__device__ __forceinline__ v16h ld16(const _Float16* p0, const _Float16* p1) {
  v8h lo = *(const v8h*)p0;
  v8h hi = *(const v8h*)p1;
  v16h r;
#pragma unroll
  for (int i = 0; i < 8; ++i) { r[i] = lo[i]; r[i + 8] = hi[i]; }
  return r;
}

// ---------------------------------------------------------------------------
// (B,C,H,W) f32 -> (B,P,C) f32
__global__ void transpose_kernel(const float* __restrict__ stem, float* __restrict__ x0) {
  int idx = blockIdx.x * 256 + threadIdx.x;   // BB*PP*CC total, launched exactly
  int c   = idx & (CC - 1);
  int pos = (idx >> 8) % PP;
  int b   = idx / (CC * PP);
  x0[idx] = stem[((size_t)b * CC + c) * PP + pos];
}

// f32 (mat,K,N) -> f16 transposed (mat,N,K): dst[(mat,n,k)] = src[(mat,k,n)]
__global__ void cvt_tr_kernel(const float* __restrict__ src, _Float16* __restrict__ dst,
                              int K, int N) {
  size_t i = (size_t)blockIdx.x * 256 + threadIdx.x;  // launched exactly nmat*K*N
  int KN  = K * N;
  int mat = (int)(i / KN);
  int rem = (int)(i % KN);
  int n   = rem / K;
  int k   = rem - n * K;
  dst[i] = (_Float16)src[(size_t)mat * KN + (size_t)k * N + n];
}

// ---------------------------------------------------------------------------
// Router: mean over tokens, logits, softmax-12, top-1.  One block per image.
__global__ void router_kernel(const float* __restrict__ xin,
                              const float* __restrict__ gw, const float* __restrict__ gb,
                              float* __restrict__ probs, int* __restrict__ topidx,
                              int* __restrict__ eidx, float* __restrict__ wgt) {
  __shared__ float mean[CC];
  __shared__ float logits[16];
  int b = blockIdx.x, t = threadIdx.x;
  float s = 0.f;
  const float* base = xin + (size_t)b * PP * CC + t;
  for (int p = 0; p < PP; ++p) s += base[(size_t)p * CC];
  mean[t] = s * (1.f / (float)PP);
  __syncthreads();
  if (t < EE) {
    float l = gb[t];
    for (int c = 0; c < CC; ++c) l += mean[c] * gw[c * EE + t];
    logits[t] = l;
  }
  __syncthreads();
  if (t == 0) {
    float mx = logits[0];
    for (int e = 1; e < EE; ++e) mx = fmaxf(mx, logits[e]);
    float den = 0.f;
    for (int e = 0; e < EE; ++e) { float p = __expf(logits[e] - mx); logits[e] = p; den += p; }
    float inv = 1.f / den;
    int best = 0; float bp = -1.f;
    for (int e = 0; e < EE; ++e) {
      float p = logits[e] * inv;
      probs[(size_t)b * EE + e] = p;
      if (p > bp) { bp = p; best = e; }
    }
    topidx[b] = best;
    eidx[b]   = best;
    wgt[b]    = bp / (bp + 1e-8f);
  }
}

// ---------------------------------------------------------------------------
// LayerNorm: one wave32 per token (8 waves / block), f32 in -> f16 out.
__global__ void ln_kernel(const float* __restrict__ xin, _Float16* __restrict__ out,
                          const float* __restrict__ g, const float* __restrict__ bb,
                          const int* __restrict__ eidx) {
  int wid  = threadIdx.x >> 5;
  int lane = threadIdx.x & 31;
  int p = blockIdx.x * 8 + wid;           // token index in [0, BB*PP)
  int b = p / PP;
  const float* row = xin + (size_t)p * CC;
  float v[8];
  float s = 0.f, ss = 0.f;
#pragma unroll
  for (int i = 0; i < 8; ++i) { v[i] = row[lane * 8 + i]; s += v[i]; ss += v[i] * v[i]; }
#pragma unroll
  for (int m = 16; m >= 1; m >>= 1) {
    s  += __shfl_xor(s,  m, 32);
    ss += __shfl_xor(ss, m, 32);
  }
  float mu  = s  * (1.f / (float)CC);
  float var = ss * (1.f / (float)CC) - mu * mu;
  float inv = rsqrtf(var + 1e-5f);
  int e = eidx[b];
  const float* gg  = g  + (size_t)e * CC;
  const float* bbp = bb + (size_t)e * CC;
  _Float16* orow = out + (size_t)p * CC;
#pragma unroll
  for (int i = 0; i < 8; ++i) {
    int c = lane * 8 + i;
    orow[c] = (_Float16)((v[i] - mu) * inv * gg[c] + bbp[c]);
  }
}

// ---------------------------------------------------------------------------
// Attention: per (image, row) block; per-row softmax over W=14; 4 heads, d=64.
__global__ void attn_kernel(const _Float16* __restrict__ q, const _Float16* __restrict__ k,
                            const _Float16* __restrict__ v, _Float16* __restrict__ o) {
  __shared__ _Float16 sk[14 * CC];
  __shared__ _Float16 sv[14 * CC];
  int b  = blockIdx.x / 14;
  int hr = blockIdx.x % 14;
  size_t rowbase = ((size_t)b * PP + hr * 14) * CC;
  for (int i = threadIdx.x; i < 14 * CC; i += 64) {
    sk[i] = k[rowbase + i];
    sv[i] = v[rowbase + i];
  }
  __syncthreads();
  int t = threadIdx.x;
  if (t < 56) {
    int nh = t / 14, wq = t % 14;
    const _Float16* qr = q + rowbase + (size_t)wq * CC + nh * 64;
    float qreg[64];
#pragma unroll
    for (int d = 0; d < 64; ++d) qreg[d] = (float)qr[d];
    float sc[14];
    float mx = -1e30f;
    for (int kw = 0; kw < 14; ++kw) {
      const _Float16* kr = sk + kw * CC + nh * 64;
      float dot = 0.f;
#pragma unroll
      for (int d = 0; d < 64; ++d) dot += qreg[d] * (float)kr[d];
      dot *= 0.125f;                       // 1/sqrt(64)
      sc[kw] = dot;
      mx = fmaxf(mx, dot);
    }
    float den = 0.f;
    for (int kw = 0; kw < 14; ++kw) { sc[kw] = __expf(sc[kw] - mx); den += sc[kw]; }
    float inv = 1.f / den;
    _Float16* orow = o + rowbase + (size_t)wq * CC + nh * 64;
#pragma unroll
    for (int d = 0; d < 64; ++d) {
      float acc = 0.f;
      for (int kw = 0; kw < 14; ++kw) acc += sc[kw] * (float)sv[kw * CC + nh * 64 + d];
      orow[d] = (_Float16)(acc * inv);
    }
  }
}

// ---------------------------------------------------------------------------
// WMMA GEMM: Y[b] = X[b](196xK) @ W[e(b)](KxN), W stored pre-transposed (N,K).
// Block tile 64x128 (4 waves, each 32x64 = 2x4 WMMA tiles), K step 32,
// double-buffered LDS, one barrier per K-step.
// MODE 0: f16 out = acc + bias                       (QKV)
// MODE 1: f32 out = acc + bias + xadd                (O proj + residual -> xres)
// MODE 2: f16 out = gelu(acc + bias)                 (MLP1)
// MODE 3: f32 out = wgt[b] * (xadd + acc + bias)     (MLP2 -> d_out layer slice)
template <int MODE, int KT, int NT>
__global__ __launch_bounds__(128)
void gemm_kernel(const _Float16* __restrict__ X, const _Float16* __restrict__ Wt,
                 const float* __restrict__ bias, const int* __restrict__ eidx,
                 const float* __restrict__ xadd, float* __restrict__ fout,
                 _Float16* __restrict__ hout, const float* __restrict__ wgt) {
  __shared__ alignas(16) _Float16 sA[2][64 * LDSK];
  __shared__ alignas(16) _Float16 sB[2][128 * LDSK];   // W^T rows: sB[n][k]
  int b  = blockIdx.z;
  int m0 = blockIdx.y * 64;
  int n0 = blockIdx.x * 128;
  int e  = eidx[b];
  const _Float16* Wm = Wt + (size_t)e * KT * NT;       // (NT, KT) row-major
  int tid  = threadIdx.x;
  int lane = tid & 31;
  int wave = tid >> 5;
  int wm = wave >> 1, wn = wave & 1;
  v8f acc[2][4] = {};

  // staging geometry
  int arow = tid >> 1;
  int aseg = (tid & 1) * 16;
  int agm  = m0 + arow; if (agm > PP - 1) agm = PP - 1;   // clamp: masked at store
  const _Float16* aSrc = X  + ((size_t)b * PP + agm) * KT + aseg;
  const _Float16* bSrc = Wm + (size_t)(n0 + tid) * KT;

  uint4 ra0, ra1, rb0, rb1, rb2, rb3;
  auto gload = [&](int ks) {
    const uint4* ap = (const uint4*)(aSrc + ks);
    ra0 = ap[0]; ra1 = ap[1];
    const uint4* bp = (const uint4*)(bSrc + ks);
    rb0 = bp[0]; rb1 = bp[1]; rb2 = bp[2]; rb3 = bp[3];
  };
  auto sstore = [&](int buf) {
    _Float16* pa = &sA[buf][arow * LDSK + aseg];
    *(uint4*)pa = ra0; *(uint4*)(pa + 8) = ra1;
    _Float16* pb = &sB[buf][tid * LDSK];
    *(uint4*)pb = rb0; *(uint4*)(pb + 8) = rb1;
    *(uint4*)(pb + 16) = rb2; *(uint4*)(pb + 24) = rb3;
  };

  int hsel = lane >> 4;            // 0: lanes 0-15, 1: lanes 16-31
  int l15  = lane & 15;
  int akb  = hsel * 8;             // A half-offset: K {0..7,16..23} vs {8..15,24..31}
  int bkb  = hsel * 8;             // B half-offset: K {0..7,16..23} vs {8..15,24..31}

  constexpr int STEPS = KT / 32;
  gload(0);
  sstore(0);
#pragma unroll
  for (int s = 0; s < STEPS; ++s) {
    if (s + 1 < STEPS) gload((s + 1) * 32);
    __syncthreads();
    int buf = s & 1;
#pragma unroll
    for (int i = 0; i < 2; ++i) {
      const _Float16* ap = &sA[buf][(wm * 32 + i * 16 + l15) * LDSK + akb];
      v16h a = ld16(ap, ap + 16);
#pragma unroll
      for (int j = 0; j < 4; ++j) {
        const _Float16* bp = &sB[buf][(wn * 64 + j * 16 + l15) * LDSK + bkb];
        v16h bf = ld16(bp, bp + 16);
        acc[i][j] = __builtin_amdgcn_wmma_f32_16x16x32_f16(
            false, a, false, bf, (short)0, acc[i][j], false, false);
      }
    }
    if (s + 1 < STEPS) sstore((s + 1) & 1);
  }

  // ---- epilogue: C/D layout: vgpr r -> row r + 8*(lane>=16); col = lane&15
  const float* bb = bias + (size_t)e * NT;
  float w = (MODE == 3) ? wgt[b] : 0.f;
  int mofs = hsel * 8;
#pragma unroll
  for (int i = 0; i < 2; ++i) {
#pragma unroll
    for (int j = 0; j < 4; ++j) {
      int ng = n0 + wn * 64 + j * 16 + l15;
      float bv = bb[ng];
#pragma unroll
      for (int r = 0; r < 8; ++r) {
        int mg = m0 + wm * 32 + i * 16 + mofs + r;
        if (mg >= PP) continue;
        size_t oi = ((size_t)b * PP + mg) * NT + ng;
        float val = acc[i][j][r] + bv;
        if (MODE == 0) {
          hout[oi] = (_Float16)val;
        } else if (MODE == 1) {
          fout[oi] = val + xadd[oi];
        } else if (MODE == 2) {
          float gv = 0.5f * val * (1.f + erff(val * 0.70710678118f));
          hout[oi] = (_Float16)gv;
        } else {
          fout[oi] = w * (xadd[oi] + val);
        }
      }
    }
  }
}

// ---------------------------------------------------------------------------
extern "C" void kernel_launch(void* const* d_in, const int* in_sizes, int n_in,
                              void* d_out, int out_size, void* d_ws, size_t ws_size,
                              hipStream_t stream) {
  (void)in_sizes; (void)n_in; (void)out_size; (void)ws_size;
  const float* stem   = (const float*)d_in[0];
  const float* gate_w = (const float*)d_in[1];
  const float* gate_b = (const float*)d_in[2];
  const float* ln1_g  = (const float*)d_in[3];
  const float* ln1_b  = (const float*)d_in[4];
  const float* wq     = (const float*)d_in[5];
  const float* bq     = (const float*)d_in[6];
  const float* wk     = (const float*)d_in[7];
  const float* bk     = (const float*)d_in[8];
  const float* wv     = (const float*)d_in[9];
  const float* bv     = (const float*)d_in[10];
  const float* wo     = (const float*)d_in[11];
  const float* bo     = (const float*)d_in[12];
  const float* ln2_g  = (const float*)d_in[13];
  const float* ln2_b  = (const float*)d_in[14];
  const float* w1     = (const float*)d_in[15];
  const float* b1     = (const float*)d_in[16];
  const float* w2     = (const float*)d_in[17];
  const float* b2     = (const float*)d_in[18];

  float* out       = (float*)d_out;
  float* out_x     = out;                                          // (4,B,196,256)
  size_t xelems    = (size_t)LL * BB * PP * CC;                    // 25,690,112
  float* out_probs = out + xelems;                                 // (4B,12)
  int*   out_idx   = (int*)(out + xelems + (size_t)LL * BB * EE);  // (4B,1)

  // ---- workspace carve-up
  char* ws = (char*)d_ws;
  auto alloc = [&](size_t bytes) {
    char* p = ws;
    ws += (bytes + 255) & ~(size_t)255;
    return p;
  };
  size_t actF32 = (size_t)BB * PP * CC * sizeof(float);
  size_t actF16 = (size_t)BB * PP * CC * sizeof(_Float16);
  float*    x0   = (float*)alloc(actF32);
  float*    xres = (float*)alloc(actF32);
  _Float16* h16  = (_Float16*)alloc(actF16);
  _Float16* q16  = (_Float16*)alloc(actF16);
  _Float16* k16  = (_Float16*)alloc(actF16);
  _Float16* v16  = (_Float16*)alloc(actF16);
  _Float16* o16  = (_Float16*)alloc(actF16);
  _Float16* m16  = (_Float16*)alloc((size_t)BB * PP * MM * sizeof(_Float16));
  int wCC = LL * EE * CC * CC;   // 3,145,728
  int wCM = LL * EE * CC * MM;   // 1,572,864
  _Float16* wq16 = (_Float16*)alloc((size_t)wCC * 2);   // all stored transposed (N,K)
  _Float16* wk16 = (_Float16*)alloc((size_t)wCC * 2);
  _Float16* wv16 = (_Float16*)alloc((size_t)wCC * 2);
  _Float16* wo16 = (_Float16*)alloc((size_t)wCC * 2);
  _Float16* w116 = (_Float16*)alloc((size_t)wCM * 2);
  _Float16* w216 = (_Float16*)alloc((size_t)wCM * 2);
  int*   eidx = (int*)alloc(BB * sizeof(int));
  float* wgt  = (float*)alloc(BB * sizeof(float));

  // ---- one-time prep (re-done every call; deterministic)
  transpose_kernel<<<(BB * PP * CC) / 256, 256, 0, stream>>>(stem, x0);
  cvt_tr_kernel<<<wCC / 256, 256, 0, stream>>>(wq, wq16, CC, CC);
  cvt_tr_kernel<<<wCC / 256, 256, 0, stream>>>(wk, wk16, CC, CC);
  cvt_tr_kernel<<<wCC / 256, 256, 0, stream>>>(wv, wv16, CC, CC);
  cvt_tr_kernel<<<wCC / 256, 256, 0, stream>>>(wo, wo16, CC, CC);
  cvt_tr_kernel<<<wCM / 256, 256, 0, stream>>>(w1, w116, CC, MM);
  cvt_tr_kernel<<<wCM / 256, 256, 0, stream>>>(w2, w216, MM, CC);

  dim3 blk(128);
  dim3 gN256(CC / 128, 4, BB);   // N=256 -> 2 n-tiles
  dim3 gN128(MM / 128, 4, BB);   // N=128 -> 1 n-tile
  int lnBlocks = (BB * PP) / 8;  // 3136

  for (int li = 0; li < LL; ++li) {
    const float* xin = (li == 0) ? x0 : (out_x + (size_t)(li - 1) * BB * PP * CC);
    size_t oE  = (size_t)li * EE;

    router_kernel<<<BB, 256, 0, stream>>>(
        xin, gate_w + (size_t)li * CC * EE, gate_b + oE,
        out_probs + oE * BB, out_idx + (size_t)li * BB, eidx, wgt);

    ln_kernel<<<lnBlocks, 256, 0, stream>>>(xin, h16,
        ln1_g + oE * CC, ln1_b + oE * CC, eidx);

    gemm_kernel<0, CC, CC><<<gN256, blk, 0, stream>>>(h16, wq16 + oE * CC * CC,
        bq + oE * CC, eidx, nullptr, nullptr, q16, nullptr);
    gemm_kernel<0, CC, CC><<<gN256, blk, 0, stream>>>(h16, wk16 + oE * CC * CC,
        bk + oE * CC, eidx, nullptr, nullptr, k16, nullptr);
    gemm_kernel<0, CC, CC><<<gN256, blk, 0, stream>>>(h16, wv16 + oE * CC * CC,
        bv + oE * CC, eidx, nullptr, nullptr, v16, nullptr);

    attn_kernel<<<BB * 14, 64, 0, stream>>>(q16, k16, v16, o16);

    gemm_kernel<1, CC, CC><<<gN256, blk, 0, stream>>>(o16, wo16 + oE * CC * CC,
        bo + oE * CC, eidx, xin, xres, nullptr, nullptr);

    ln_kernel<<<lnBlocks, 256, 0, stream>>>(xres, h16,
        ln2_g + oE * CC, ln2_b + oE * CC, eidx);

    gemm_kernel<2, CC, MM><<<gN128, blk, 0, stream>>>(h16, w116 + oE * CC * MM,
        b1 + oE * MM, eidx, nullptr, nullptr, m16, nullptr);

    gemm_kernel<3, MM, CC><<<gN256, blk, 0, stream>>>(m16, w216 + oE * MM * CC,
        b2 + oE * CC, eidx, xres, out_x + (size_t)li * BB * PP * CC, nullptr, wgt);
  }
}
